// SphericalResCNN_4698694222687
// MI455X (gfx1250) — matrix operand order repaired and checked by reading, not compiled, not run on hardware
//
#include <hip/hip_runtime.h>
#include <cstddef>

typedef _Float16 half_t;
typedef __attribute__((ext_vector_type(16))) _Float16 v16h;
typedef __attribute__((ext_vector_type(8)))  float    v8f;

// ---------------- shared helpers ----------------
__host__ __device__ inline int imin_(int a, int b) { return a < b ? a : b; }
__host__ __device__ inline int imax_(int a, int b) { return a > b ? a : b; }

// Canonical CG-table offset (global enumeration for lmax=8).
__host__ __device__ inline int cg_offset(int L1, int L2, int L) {
  int off = 0;
  for (int a1 = 0; a1 <= 8; ++a1)
    for (int a2 = 0; a2 <= a1; ++a2) {
      int hi = imin_(a1 + a2, 8);
      for (int al = a1 - a2; al <= hi; ++al) {
        if (a1 == L1 && a2 == L2 && al == L) return off;
        off += (2 * al + 1) * (2 * a1 + 1) * (2 * a2 + 1);
      }
    }
  return off;
}

// ---------------- workspace layout (bytes, 256B aligned) ----------------
constexpr size_t al256(size_t x) { return (x + 255) & ~(size_t)255; }
static constexpr size_t CG_FLOATS   = 320000;            // actual need: 259107
static constexpr size_t OFF_CG      = 0;
static constexpr size_t OFF_FNORM   = al256(OFF_CG + CG_FLOATS * 4);
static constexpr size_t OFF_STD     = al256(OFF_FNORM + 16 * 81 * 2 * 4);
static constexpr size_t OFF_FS00    = al256(OFF_STD + 64);
static constexpr size_t OFF_LOUT    = al256(OFF_FS00 + 128);
static constexpr size_t LOUT_FLOATS = 16 * 16 * 81 * 2;              // per-layer out
static constexpr size_t OFF_FIN     = al256(OFF_LOUT + 6 * LOUT_FLOATS * 4);
static constexpr int    FIN_LSTRIDE = 16 * 17 * 17 * 2;              // floats per l
static constexpr size_t FIN_FLOATS  = 9 * (size_t)FIN_LSTRIDE;
static constexpr size_t OFF_A       = al256(OFF_FIN + FIN_FLOATS * 4);
static constexpr size_t A_HALVES    = 16 * 18496;                    // Kmax cap (64-pad)
static constexpr size_t OFF_X       = al256(OFF_A + A_HALVES * 2);   // X is last (headroom)

struct TripleRec { int l1, l2, t1, t2, colOff, cgOff; };
struct TripleTab { TripleRec r[32]; };

// ---------------- CG coefficient table (double precision, once per call) ----------------
__device__ inline double dfact_(int n) {
  double r = 1.0;
  for (int i = 2; i <= n; ++i) r *= (double)i;
  return r;
}

__device__ float cg_coef_dev(int l1, int m1, int l2, int m2, int l, int m) {
  if (m1 + m2 != m) return 0.f;
  double pref = (2.0 * l + 1.0) * dfact_(l + l1 - l2) * dfact_(l - l1 + l2) *
                dfact_(l1 + l2 - l) / dfact_(l1 + l2 + l + 1);
  pref = sqrt(pref * dfact_(l + m) * dfact_(l - m) * dfact_(l1 - m1) *
              dfact_(l1 + m1) * dfact_(l2 - m2) * dfact_(l2 + m2));
  int kmin = imax_(0, imax_(l2 - l - m1, l1 + m2 - l));
  int kmax = imin_(l1 + l2 - l, imin_(l1 - m1, l2 + m2));
  double s = 0.0;
  for (int k = kmin; k <= kmax; ++k) {
    double d = dfact_(k) * dfact_(l1 + l2 - l - k) * dfact_(l1 - m1 - k) *
               dfact_(l2 + m2 - k) * dfact_(l - l2 + m1 + k) * dfact_(l - l1 - m2 + k);
    s += ((k & 1) ? -1.0 : 1.0) / d;
  }
  return (float)(pref * s);
}

__global__ void srcnn_cg_init(float* __restrict__ cg) {
  int id = blockIdx.x;                 // 9*9*9 grid over (l1,l2,l)
  int l1 = id / 81, l2 = (id / 9) % 9, l = id % 9;
  if (l2 > l1) return;
  if (l < l1 - l2 || l > imin_(l1 + l2, 8)) return;
  int off = cg_offset(l1, l2, l);
  int n1 = 2 * l1 + 1, n2 = 2 * l2 + 1, nm = 2 * l + 1;
  int total = nm * n1 * n2;
  for (int idx = threadIdx.x; idx < total; idx += blockDim.x) {
    int bb = idx % n2;
    int a  = (idx / n2) % n1;
    int mm = idx / (n1 * n2);
    cg[off + idx] = cg_coef_dev(l1, a - l1, l2, bb - l2, l, mm - l);
  }
}

// ---------------- chop: std reduction + normalize ----------------
__global__ void srcnn_chop_std(const float* __restrict__ src, float* __restrict__ stdOut) {
  __shared__ float ssum[512];
  __shared__ float ssq[512];
  float s = 0.f, q = 0.f;
  for (int i = threadIdx.x; i < 16 * 81; i += blockDim.x) {
    float re = src[i * 2], im = src[i * 2 + 1];
    float m = sqrtf(re * re + im * im);
    s += m; q += m * m;
  }
  ssum[threadIdx.x] = s; ssq[threadIdx.x] = q;
  __syncthreads();
  for (int off = blockDim.x >> 1; off > 0; off >>= 1) {
    if ((int)threadIdx.x < off) {
      ssum[threadIdx.x] += ssum[threadIdx.x + off];
      ssq[threadIdx.x]  += ssq[threadIdx.x + off];
    }
    __syncthreads();
  }
  if (threadIdx.x == 0) {
    float n = 16.f * 81.f;
    float mean = ssum[0] / n;
    float var = ssq[0] / n - mean * mean;
    stdOut[0] = sqrtf(fmaxf(var, 0.f));
  }
}

__global__ void srcnn_chop_norm(const float* __restrict__ src, float* __restrict__ dst,
                                const float* __restrict__ stdv, float* __restrict__ fs00,
                                int save0) {
  int idx = blockIdx.x * blockDim.x + threadIdx.x;
  if (idx >= 16 * 81 * 2) return;
  float s = fmaxf(1e-4f, stdv[0]);
  float v = src[idx] / s;
  dst[idx] = v;
  if (save0) {
    int ri = idx & 1;
    int m = (idx >> 1) % 81;
    int b = idx / (81 * 2);
    if (m == 0) fs00[b * 2 + ri] = v;
  }
}

// ---------------- gather layer inputs into f_in[l][b][ch][m][ri] ----------------
__global__ void srcnn_build_fin(const float* __restrict__ fnorm,
                                const float* __restrict__ prevLout,
                                float* __restrict__ fin,
                                int lmax, int prevLmax, int mode) {
  int idx = blockIdx.x * blockDim.x + threadIdx.x;
  if (idx >= 9 * 16 * 17 * 17 * 2) return;
  int ri = idx & 1;
  int r = idx >> 1;
  int m = r % 17;  r /= 17;
  int ch = r % 17; r /= 17;
  int b = r % 16;
  int l = r / 16;
  float v = 0.f;
  if (l <= lmax && m < 2 * l + 1) {
    int gm = l * l + m;
    if (mode == 1) {                              // mid-step layer: input = prev out (tau 16)
      if (ch < 16) v = prevLout[((size_t)(b * 16 + ch) * 81 + gm) * 2 + ri];
    } else {                                      // step head: [chop(1ch) | prev out(16ch)]
      if (ch == 0) v = fnorm[((size_t)b * 81 + gm) * 2 + ri];
      else if (l <= prevLmax) v = prevLout[((size_t)(b * 16 + (ch - 1)) * 81 + gm) * 2 + ri];
    }
  }
  fin[(size_t)l * FIN_LSTRIDE + ((size_t)(b * 17 + ch) * 17 + m) * 2 + ri] = v;
}

// ---------------- CG tensor product -> realified, k-tiled B-matrix X (f16) ----------------
// X layout: [b][kt=k/32][n=0..Npad)[kk=k%32] halves -> lane fragment is 16 contiguous
// halves at (...*Npad + n)*32 + hi*16.  Column c occupies rows 2c (=[xr|xi]) and
// 2c+1 (=[-xi|xr]); both rows live in the same 32-k tile since 2c is even.
__global__ __launch_bounds__(128) void srcnn_cg(
    const float* __restrict__ fin, const float* __restrict__ cg,
    half_t* __restrict__ X, TripleTab tab, int ntr, int l, int Kp, int Npad) {
  int tr = blockIdx.x >> 4;
  int b  = blockIdx.x & 15;
  const TripleRec R = tab.r[tr];
  const int n1 = 2 * R.l1 + 1, n2 = 2 * R.l2 + 1, nm = 2 * l + 1;
  const int nkt = Kp >> 5;
  const float* f1b = fin + (size_t)R.l1 * FIN_LSTRIDE + (size_t)b * (17 * 17 * 2);
  const float* f2b = fin + (size_t)R.l2 * FIN_LSTRIDE + (size_t)b * (17 * 17 * 2);
  const float* Cbase = cg + R.cgOff;
  int ncol = R.t1 * R.t2;
  for (int col = threadIdx.x; col < ncol; col += blockDim.x) {
    int i = col / R.t2, j = col - i * R.t2;
    const float* f1 = f1b + (size_t)i * (17 * 2);
    const float* f2 = f2b + (size_t)j * (17 * 2);
    int k0 = 2 * (R.colOff + col);
    half_t* tile = X + ((size_t)(b * nkt + (k0 >> 5)) * Npad) * 32;
    int kk = k0 & 31;
    for (int mi = 0; mi < nm; ++mi) {
      float xr = 0.f, xi = 0.f;
      int mrel = mi - l;
      for (int a = 0; a < n1; ++a) {
        int bb = mrel - (a - R.l1) + R.l2;       // CG sparsity: m1+m2 == m
        if ((unsigned)bb >= (unsigned)n2) continue;
        float c = Cbase[(mi * n1 + a) * n2 + bb];
        float ar = f1[a * 2], ai = f1[a * 2 + 1];
        float br = f2[bb * 2], bi = f2[bb * 2 + 1];
        xr += c * (ar * br - ai * bi);
        xi += c * (ar * bi + ai * br);
      }
      tile[(size_t)mi * 32 + kk]            = (half_t)xr;
      tile[(size_t)(nm + mi) * 32 + kk]     = (half_t)xi;
      tile[(size_t)mi * 32 + kk + 1]        = (half_t)(-xi);
      tile[(size_t)(nm + mi) * 32 + kk + 1] = (half_t)xr;
    }
  }
}

// ---------------- weight prep: swizzled A fragments ----------------
// A2 layout: [kt][t][hi][h=0..16) halves = exact per-lane WMMA A-fragment, so the
// GEMM loads one contiguous v16h per lane per k-tile.
// Fragment half h of lane (t,hi) holds original k = kt*32 + hi*16 + h.
__global__ void srcnn_build_a(const float* __restrict__ w, const float* __restrict__ sc,
                              half_t* __restrict__ A2, int C, int Kp) {
  int idx = blockIdx.x * blockDim.x + threadIdx.x;
  if (idx >= 16 * Kp) return;
  int t = idx / Kp, k = idx - t * Kp;
  int c = k >> 1;
  half_t v = (half_t)0.f;
  if (c < C) {
    float s = fmaxf(1e-5f, sc[c]);
    v = (half_t)(w[((size_t)t * C + c) * 2 + (k & 1)] / s);
  }
  int kt = k >> 5, kk = k & 31;
  int hi = kk >> 4;
  int h  = kk & 15;
  A2[(((size_t)kt * 16 + t) * 2 + hi) * 16 + h] = v;
}

// ---------------- WMMA GEMM: Out[16, 2(2l+1)] = A[16,Kp] x X[Kp,Npad] ----------------
__global__ __launch_bounds__(32) void srcnn_gemm(
    const half_t* __restrict__ A2, const half_t* __restrict__ X,
    float* __restrict__ outBase, int Kp, int Npad, int l) {
  int b = blockIdx.x;
  int ntile = blockIdx.y;
  int ln = threadIdx.x;
  int t = ln & 15;
  int hi = ln >> 4;
  int n = ntile * 16 + t;              // this lane's D/B column
  int nkt = Kp >> 5;                   // even (Kp is a multiple of 64)
  const half_t* Ap = A2 + ((size_t)t * 2 + hi) * 16;
  const half_t* Xp = X + ((size_t)(b * nkt) * Npad + n) * 32 + hi * 16;
  const size_t aStep = 16 * 2 * 16;    // halves per k-tile
  const size_t xStep = (size_t)Npad * 32;
  v8f acc = {};
  for (int kt = 0; kt < nkt; kt += 2) {
    v16h a0 = *(const v16h*)(Ap);
    v16h b0 = *(const v16h*)(Xp);
    v16h a1 = *(const v16h*)(Ap + aStep);
    v16h b1 = *(const v16h*)(Xp + xStep);
    if (kt + 2 < nkt)
      __builtin_prefetch(Xp + 2 * xStep, 0, 1);
    acc = __builtin_amdgcn_wmma_f32_16x16x32_f16(false, a0, false, b0,
                                                 (short)0, acc, false, false);
    acc = __builtin_amdgcn_wmma_f32_16x16x32_f16(false, a1, false, b1,
                                                 (short)0, acc, false, false);
    Ap += 2 * aStep;
    Xp += 2 * xStep;
  }
  int M2 = 2 * (2 * l + 1);
  if (n < M2) {
    int m, ri;
    if (n < 2 * l + 1) { m = n; ri = 0; }
    else               { m = n - (2 * l + 1); ri = 1; }
#pragma unroll
    for (int r = 0; r < 8; ++r) {       // D: lane col=ln%16, rows r+8*hi
      int trow = r + (hi ? 8 : 0);
      outBase[(((size_t)b * 16 + trow) * 81 + (size_t)(l * l) + m) * 2 + ri] = acc[r];
    }
  }
}

// ---------------- final skip-connection gather ----------------
__global__ void srcnn_output(const float* __restrict__ fs00,
                             const float* __restrict__ loutBase, int loutStride,
                             float* __restrict__ out) {
  int idx = blockIdx.x * blockDim.x + threadIdx.x;
  if (idx >= 16 * 194) return;
  int b = idx / 194, p = idx - b * 194;
  float v;
  if (p < 2) v = fs00[b * 2 + p];
  else {
    int q = p - 2;
    int u = q / 32, r = q % 32, t = r >> 1, ri = r & 1;
    v = loutBase[(size_t)u * loutStride + ((size_t)(b * 16 + t) * 81 + 0) * 2 + ri];
  }
  out[idx] = v;
}

// ---------------- host orchestration ----------------
static const int kLmaxU[6] = {4, 4, 6, 6, 8, 8};
static const int kWsPrefix[6] = {0, 5, 10, 17, 24, 33};   // 5,5,7,7,9,9 arrays per layer

static inline int tauIn(int u, int l) {
  switch (u) {
    case 0: return 1;
    case 1: return 16;
    case 2: return (l <= 4) ? 17 : 1;
    case 3: return 16;
    case 4: return (l <= 6) ? 17 : 1;
    case 5: return 16;
  }
  return 0;
}

extern "C" void kernel_launch(void* const* d_in, const int* in_sizes, int n_in,
                              void* d_out, int out_size, void* d_ws, size_t ws_size,
                              hipStream_t stream) {
  (void)in_sizes; (void)n_in; (void)out_size; (void)ws_size;
  char* ws = (char*)d_ws;
  float*  cg       = (float*)(ws + OFF_CG);
  float*  fnorm    = (float*)(ws + OFF_FNORM);
  float*  stdv     = (float*)(ws + OFF_STD);
  float*  fs00     = (float*)(ws + OFF_FS00);
  float*  loutBase = (float*)(ws + OFF_LOUT);
  float*  fin      = (float*)(ws + OFF_FIN);
  half_t* Abuf     = (half_t*)(ws + OFF_A);
  half_t* Xbuf     = (half_t*)(ws + OFF_X);
  const float* f0  = (const float*)d_in[0];

  srcnn_cg_init<<<729, 256, 0, stream>>>(cg);

  for (int u = 0; u < 6; ++u) {
    int lmax = kLmaxU[u];
    float* lout = loutBase + (size_t)u * LOUT_FLOATS;
    if ((u & 1) == 0) {                              // step head: re-chop the input
      const float* src = (u == 0) ? f0 : fnorm;
      srcnn_chop_std<<<1, 512, 0, stream>>>(src, stdv);
      srcnn_chop_norm<<<(16 * 81 * 2 + 255) / 256, 256, 0, stream>>>(
          src, fnorm, stdv, fs00, (u == 0) ? 1 : 0);
    }
    int prevLmax = (u == 0) ? -1 : kLmaxU[u - 1];
    const float* prevLout = (u == 0) ? nullptr
                                     : (loutBase + (size_t)(u - 1) * LOUT_FLOATS);
    srcnn_build_fin<<<(9 * 16 * 17 * 17 * 2 + 255) / 256, 256, 0, stream>>>(
        fnorm, prevLout, fin, lmax, prevLmax, (u & 1));

    for (int l = 0; l <= lmax; ++l) {
      TripleTab tab = {};
      int ntr = 0, ncols = 0;
      for (int l1 = 0; l1 <= lmax; ++l1)
        for (int l2 = 0; l2 <= l1; ++l2) {
          if (l < l1 - l2 || l > imin_(l1 + l2, lmax)) continue;
          TripleRec& R = tab.r[ntr++];
          R.l1 = l1; R.l2 = l2;
          R.t1 = tauIn(u, l1); R.t2 = tauIn(u, l2);
          R.colOff = ncols;
          R.cgOff = cg_offset(l1, l2, l);
          ncols += R.t1 * R.t2;
        }
      int K = 2 * ncols;
      int Kp = (K + 63) & ~63;                       // multiple of 64 (2 k-tiles)
      int M2 = 2 * (2 * l + 1);
      int Npad = (M2 + 15) & ~15;

      srcnn_cg<<<dim3(ntr * 16), 128, 0, stream>>>(fin, cg, Xbuf, tab, ntr, l, Kp, Npad);

      const float* wptr  = (const float*)d_in[1 + kWsPrefix[u] + l];
      const float* scptr = (const float*)d_in[43 + kWsPrefix[u] + l];
      srcnn_build_a<<<(16 * Kp + 255) / 256, 256, 0, stream>>>(wptr, scptr, Abuf, ncols, Kp);

      srcnn_gemm<<<dim3(16, Npad / 16), 32, 0, stream>>>(Abuf, Xbuf, lout, Kp, Npad, l);
    }
  }
  srcnn_output<<<(16 * 194 + 255) / 256, 256, 0, stream>>>(
      fs00, loutBase, (int)LOUT_FLOATS, (float*)d_out);
}
